// GNN_h_46428596469875
// MI455X (gfx1250) — matrix operand order repaired and checked by reading, not compiled, not run on hardware
//
#include <hip/hip_runtime.h>
#include <hip/hip_bf16.h>
#include <cmath>

// ---------------------------------------------------------------------------
// MI455X GNN message-passing step, v2.
// Edge MLPs dominate (~69 GFLOP); z_h (12.8MB) is L2-resident, so the kernel
// is matrix-op bound -> v_wmma_f32_16x16x32_f16 everywhere.
// v2: 64-row M-blocking (4 x 16 subtiles per wave): each packed B fragment is
// loaded once and fed to 4 WMMAs -> ~4x better wmma:vmem ratio. The gate's
// final 128->1 layer also runs on WMMA via a zero-padded 16-column B panel.
// One wave32 per block; single-wave workgroups (barriers lower to S_NOP).
// ---------------------------------------------------------------------------

typedef __attribute__((ext_vector_type(16))) _Float16 v16h;
typedef __attribute__((ext_vector_type(8)))  float    v8f;

union AF { v16h v; unsigned int u[8]; };
union CF { v8f  v; float f[8]; };

#define MT 4  // M subtiles per wave (4 x 16 = 64 rows)

// A fragment (16x32 f16, MxK) from an LDS row-major tile.
// ISA layout: lane = half*16 + M; VGPR v holds K pair:
//   k = kchunk*32 + (v>=4?16:0) + half*8 + (v%4)*2
__device__ static inline v16h lds_load_a(const _Float16* base, int strideDW,
                                         int kchunk, int lane) {
  AF a;
  const unsigned int* b = (const unsigned int*)base;
  const int row  = lane & 15;
  const int half = lane >> 4;
  const int rowOff = row * strideDW;
#pragma unroll
  for (int v = 0; v < 8; ++v) {
    int k = kchunk * 32 + ((v >> 2) << 4) + half * 8 + ((v & 3) << 1);
    a.u[v] = b[rowOff + (k >> 1)];
  }
  return a.v;
}

// B fragment (32x16 f16, KxN) from packed global weights.
// Packed layout: dword index = (k/2)*Ncols + n, halves = W[k][n], W[k+1][n].
__device__ static inline v16h glob_load_b(const _Float16* Wp, int Ncols,
                                          int kchunk, int ntile, int lane) {
  AF a;
  const unsigned int* b = (const unsigned int*)Wp;
  const int ncol  = ntile * 16 + (lane & 15);
  const int pbase = (kchunk * 32 + (lane >> 4) * 8) >> 1;
#pragma unroll
  for (int v = 0; v < 8; ++v) {
    a.u[v] = b[(pbase + ((v >> 2) << 3) + (v & 3)) * Ncols + ncol];
  }
  return a.v;
}

__device__ static inline v8f wmma32(v16h a, v16h b, v8f c) {
  return __builtin_amdgcn_wmma_f32_16x16x32_f16(false, a, false, b,
                                                (short)0, c, false, false);
}

__device__ static inline int imin(int a, int b) { return a < b ? a : b; }

// ---------------------------------------------------------------------------
// Weight packing: fp32 [Kreal x Ncols] row-major -> f16 packed, K padded.
// ---------------------------------------------------------------------------
__global__ void gnn_pack_w(const float* __restrict__ src, _Float16* __restrict__ dst,
                           int Kreal, int Kpad, int Ncols) {
  int i = blockIdx.x * blockDim.x + threadIdx.x;
  if (i >= Kpad * Ncols) return;
  int k = i / Ncols, n = i % Ncols;
  float v = (k < Kreal) ? src[(size_t)k * Ncols + n] : 0.0f;
  dst[((size_t)(k >> 1) * Ncols + n) * 2 + (k & 1)] = (_Float16)v;
}

// Pack a [K x 1] vector into a [K x 16] panel with cols 1..15 zeroed.
__global__ void gnn_pack_wvec(const float* __restrict__ src,
                              _Float16* __restrict__ dst, int K) {
  int i = blockIdx.x * blockDim.x + threadIdx.x;
  if (i >= K * 16) return;
  int k = i / 16, n = i % 16;
  float v = (n == 0) ? src[k] : 0.0f;
  dst[((size_t)(k >> 1) * 16 + n) * 2 + (k & 1)] = (_Float16)v;
}

__global__ void gnn_zero_f32(float* __restrict__ p, int n) {
  for (int i = blockIdx.x * blockDim.x + threadIdx.x; i < n;
       i += gridDim.x * blockDim.x)
    p[i] = 0.0f;
}

// ---------------------------------------------------------------------------
// Edge kernel: one wave32 per 64-edge tile (4 x 16-row subtiles).
// gate: A@Wg1->relu->Ht, Ht@Wg2panel -> w[64] (sigmoid)
// msg:  A@We1->relu->Ht, Ht@We2 -> scale by w -> fp32 atomic scatter to m_agg
// ---------------------------------------------------------------------------
__global__ __launch_bounds__(32) void gnn_edge_kernel(
    const float* __restrict__ z_h, const int* __restrict__ ei,
    const _Float16* __restrict__ We1p, const float* __restrict__ be1,
    const _Float16* __restrict__ We2p, const float* __restrict__ be2,
    const _Float16* __restrict__ Wg1p, const float* __restrict__ bg1,
    const _Float16* __restrict__ Wg2p, const float* __restrict__ bg2,
    float* __restrict__ m_agg, int E) {
  __shared__ _Float16 At[64 * 160];
  __shared__ _Float16 Ht[64 * 128];
  __shared__ int sidx[64], tidx[64];
  __shared__ float wg[64];

  const int lane = threadIdx.x;
  const int e0 = blockIdx.x * 64;
  const int half = lane >> 4;
  const int nl = lane & 15;

  for (int i = lane; i < 64; i += 32) {
    int e = imin(e0 + i, E - 1);
    sidx[i] = ei[e];
    tidx[i] = ei[E + e];
  }
  __syncthreads();

  // Gather zs|zt (f32 -> f16); each lane covers 2 columns per row.
  for (int r = 0; r < 64; ++r) {
    const float2 zs = ((const float2*)(z_h + (size_t)sidx[r] * 64))[lane];
    const float2 zt = ((const float2*)(z_h + (size_t)tidx[r] * 64))[lane];
    At[r * 160 + 2 * lane]          = (_Float16)zs.x;
    At[r * 160 + 2 * lane + 1]      = (_Float16)zs.y;
    At[r * 160 + 64 + 2 * lane]     = (_Float16)zt.x;
    At[r * 160 + 64 + 2 * lane + 1] = (_Float16)zt.y;
  }
  // Zero the K padding (cols 136..159).
  for (int i = lane; i < 64 * 24; i += 32)
    At[(i / 24) * 160 + 136 + (i % 24)] = (_Float16)0.0f;
  // Edge features (cols 128..135), 2 edges per lane.
  for (int e = lane; e < 64; e += 32) {
    const float* zs = z_h + (size_t)sidx[e] * 64;
    const float* zt = z_h + (size_t)tidx[e] * 64;
    float d0 = zs[0] - zt[0], d1 = zs[1] - zt[1], d2 = zs[2] - zt[2];
    float dist = d0 * d0 + d1 * d1 + d2 * d2;
    float ax = zs[3], ay = zs[4], az = zs[5];
    float bx = zt[3], by = zt[4], bz = zt[5];
    float cx = ay * bz - az * by;
    float cy = az * bx - ax * bz;
    float cz = ax * by - ay * bx;
    float cn = sqrtf(cx * cx + cy * cy + cz * cz);
    _Float16* r = At + e * 160 + 128;
    r[0] = (_Float16)d0; r[1] = (_Float16)d1; r[2] = (_Float16)d2;
    r[3] = (_Float16)dist;
    r[4] = (_Float16)cx; r[5] = (_Float16)cy; r[6] = (_Float16)cz;
    r[7] = (_Float16)cn;
  }
  __syncthreads();

  // Gate layer 1: Ht = relu(A @ Wg1 + bg1). B fragment shared by 4 subtiles.
  for (int nt = 0; nt < 8; ++nt) {
    CF c[MT];
#pragma unroll
    for (int mt = 0; mt < MT; ++mt)
#pragma unroll
      for (int v = 0; v < 8; ++v) c[mt].f[v] = 0.0f;
#pragma unroll
    for (int kc = 0; kc < 5; ++kc) {
      v16h b = glob_load_b(Wg1p, 128, kc, nt, lane);
#pragma unroll
      for (int mt = 0; mt < MT; ++mt)
        c[mt].v = wmma32(lds_load_a(At + mt * 16 * 160, 80, kc, lane), b, c[mt].v);
    }
    const int nc = nt * 16 + nl;
    const float bias = bg1[nc];
#pragma unroll
    for (int mt = 0; mt < MT; ++mt)
#pragma unroll
      for (int v = 0; v < 8; ++v) {
        float x = c[mt].f[v] + bias;
        Ht[(mt * 16 + v + half * 8) * 128 + nc] = (_Float16)fmaxf(x, 0.0f);
      }
  }
  __syncthreads();

  // Gate layer 2 on WMMA: Ht @ Wg2panel (col 0 = Wg2). Lanes with N==0 hold it.
  {
    CF c[MT];
#pragma unroll
    for (int mt = 0; mt < MT; ++mt)
#pragma unroll
      for (int v = 0; v < 8; ++v) c[mt].f[v] = 0.0f;
#pragma unroll
    for (int kc = 0; kc < 4; ++kc) {
      v16h b = glob_load_b(Wg2p, 16, kc, 0, lane);
#pragma unroll
      for (int mt = 0; mt < MT; ++mt)
        c[mt].v = wmma32(lds_load_a(Ht + mt * 16 * 128, 64, kc, lane), b, c[mt].v);
    }
    if (nl == 0) {
      const float b2 = bg2[0];
#pragma unroll
      for (int mt = 0; mt < MT; ++mt)
#pragma unroll
        for (int v = 0; v < 8; ++v) {
          int m = mt * 16 + v + half * 8;
          float x = c[mt].f[v] + b2;
          wg[m] = 1.0f / (1.0f + __expf(-x));
        }
    }
  }
  __syncthreads();

  // Message layer 1: Ht = relu(A @ We1 + be1) (overwrites gate hidden).
  for (int nt = 0; nt < 8; ++nt) {
    CF c[MT];
#pragma unroll
    for (int mt = 0; mt < MT; ++mt)
#pragma unroll
      for (int v = 0; v < 8; ++v) c[mt].f[v] = 0.0f;
#pragma unroll
    for (int kc = 0; kc < 5; ++kc) {
      v16h b = glob_load_b(We1p, 128, kc, nt, lane);
#pragma unroll
      for (int mt = 0; mt < MT; ++mt)
        c[mt].v = wmma32(lds_load_a(At + mt * 16 * 160, 80, kc, lane), b, c[mt].v);
    }
    const int nc = nt * 16 + nl;
    const float bias = be1[nc];
#pragma unroll
    for (int mt = 0; mt < MT; ++mt)
#pragma unroll
      for (int v = 0; v < 8; ++v) {
        float x = c[mt].f[v] + bias;
        Ht[(mt * 16 + v + half * 8) * 128 + nc] = (_Float16)fmaxf(x, 0.0f);
      }
  }
  __syncthreads();

  // Message layer 2 + immediate weighted scatter.
  for (int nt = 0; nt < 4; ++nt) {
    CF c[MT];
#pragma unroll
    for (int mt = 0; mt < MT; ++mt)
#pragma unroll
      for (int v = 0; v < 8; ++v) c[mt].f[v] = 0.0f;
#pragma unroll
    for (int kc = 0; kc < 4; ++kc) {
      v16h b = glob_load_b(We2p, 64, kc, nt, lane);
#pragma unroll
      for (int mt = 0; mt < MT; ++mt)
        c[mt].v = wmma32(lds_load_a(Ht + mt * 16 * 128, 64, kc, lane), b, c[mt].v);
    }
    const int nc = nt * 16 + nl;
    const float bias = be2[nc];
#pragma unroll
    for (int mt = 0; mt < MT; ++mt)
#pragma unroll
      for (int v = 0; v < 8; ++v) {
        int m = mt * 16 + v + half * 8;
        float val = (c[mt].f[v] + bias) * wg[m];
        atomicAdd(&m_agg[(size_t)tidx[m] * 64 + nc], val);
      }
  }
}

// ---------------------------------------------------------------------------
// World kernel: 64-node tiles; [64 x 96] (z|diff|pad) -> 128 -> 64; per-lane
// partial column sums (32 rows each) -> one fp32 atomic per lane per N-tile.
// ---------------------------------------------------------------------------
__global__ __launch_bounds__(32) void gnn_world_kernel(
    const float* __restrict__ z_h, const float* __restrict__ posw,
    const _Float16* __restrict__ Ww1p, const float* __restrict__ bw1,
    const _Float16* __restrict__ Ww2p, const float* __restrict__ bw2,
    float* __restrict__ wsum, int N) {
  __shared__ _Float16 At[64 * 96];
  __shared__ _Float16 Ht[64 * 128];
  const int lane = threadIdx.x;
  const int n0 = blockIdx.x * 64;
  const int half = lane >> 4;
  const int nl = lane & 15;

  for (int r = 0; r < 64; ++r) {
    int node = imin(n0 + r, N - 1);
    const float2 z = ((const float2*)(z_h + (size_t)node * 64))[lane];
    At[r * 96 + 2 * lane]     = (_Float16)z.x;
    At[r * 96 + 2 * lane + 1] = (_Float16)z.y;
  }
  for (int r = lane; r < 64; r += 32) {
    int node = imin(n0 + r, N - 1);
    const float* z = z_h + (size_t)node * 64;
    At[r * 96 + 64] = (_Float16)(z[0] - posw[0]);
    At[r * 96 + 65] = (_Float16)(z[1] - posw[1]);
    At[r * 96 + 66] = (_Float16)(z[2] - posw[2]);
  }
  for (int i = lane; i < 64 * 29; i += 32)
    At[(i / 29) * 96 + 67 + (i % 29)] = (_Float16)0.0f;
  __syncthreads();

  for (int nt = 0; nt < 8; ++nt) {
    CF c[MT];
#pragma unroll
    for (int mt = 0; mt < MT; ++mt)
#pragma unroll
      for (int v = 0; v < 8; ++v) c[mt].f[v] = 0.0f;
#pragma unroll
    for (int kc = 0; kc < 3; ++kc) {
      v16h b = glob_load_b(Ww1p, 128, kc, nt, lane);
#pragma unroll
      for (int mt = 0; mt < MT; ++mt)
        c[mt].v = wmma32(lds_load_a(At + mt * 16 * 96, 48, kc, lane), b, c[mt].v);
    }
    const int nc = nt * 16 + nl;
    const float bias = bw1[nc];
#pragma unroll
    for (int mt = 0; mt < MT; ++mt)
#pragma unroll
      for (int v = 0; v < 8; ++v) {
        float x = c[mt].f[v] + bias;
        Ht[(mt * 16 + v + half * 8) * 128 + nc] = (_Float16)fmaxf(x, 0.0f);
      }
  }
  __syncthreads();

  for (int nt = 0; nt < 4; ++nt) {
    CF c[MT];
#pragma unroll
    for (int mt = 0; mt < MT; ++mt)
#pragma unroll
      for (int v = 0; v < 8; ++v) c[mt].f[v] = 0.0f;
#pragma unroll
    for (int kc = 0; kc < 4; ++kc) {
      v16h b = glob_load_b(Ww2p, 64, kc, nt, lane);
#pragma unroll
      for (int mt = 0; mt < MT; ++mt)
        c[mt].v = wmma32(lds_load_a(Ht + mt * 16 * 128, 64, kc, lane), b, c[mt].v);
    }
    const int nc = nt * 16 + nl;
    const float bias = bw2[nc];
    float s = 0.0f;
#pragma unroll
    for (int mt = 0; mt < MT; ++mt)
#pragma unroll
      for (int v = 0; v < 8; ++v)
        if (n0 + mt * 16 + v + half * 8 < N) s += c[mt].f[v] + bias;
    atomicAdd(&wsum[nc], s);
  }
}

// ---------------------------------------------------------------------------
// Node kernel: 64-node tiles; [64 x 128] (z | m_agg+wsum) -> 128 -> 64 -> out.
// ---------------------------------------------------------------------------
__global__ __launch_bounds__(32) void gnn_node_kernel(
    const float* __restrict__ z_h, const float* __restrict__ m_agg,
    const float* __restrict__ wsum,
    const _Float16* __restrict__ Wn1p, const float* __restrict__ bn1,
    const _Float16* __restrict__ Wn2p, const float* __restrict__ bn2,
    float* __restrict__ out, int N) {
  __shared__ _Float16 At[64 * 128];
  __shared__ _Float16 Ht[64 * 128];
  const int lane = threadIdx.x;
  const int n0 = blockIdx.x * 64;
  const int half = lane >> 4;
  const int nl = lane & 15;

  const float2 ws2 = ((const float2*)wsum)[lane];
  for (int r = 0; r < 64; ++r) {
    int node = imin(n0 + r, N - 1);
    const float2 z  = ((const float2*)(z_h   + (size_t)node * 64))[lane];
    const float2 mg = ((const float2*)(m_agg + (size_t)node * 64))[lane];
    At[r * 128 + 2 * lane]          = (_Float16)z.x;
    At[r * 128 + 2 * lane + 1]      = (_Float16)z.y;
    At[r * 128 + 64 + 2 * lane]     = (_Float16)(mg.x + ws2.x);
    At[r * 128 + 64 + 2 * lane + 1] = (_Float16)(mg.y + ws2.y);
  }
  __syncthreads();

  for (int nt = 0; nt < 8; ++nt) {
    CF c[MT];
#pragma unroll
    for (int mt = 0; mt < MT; ++mt)
#pragma unroll
      for (int v = 0; v < 8; ++v) c[mt].f[v] = 0.0f;
#pragma unroll
    for (int kc = 0; kc < 4; ++kc) {
      v16h b = glob_load_b(Wn1p, 128, kc, nt, lane);
#pragma unroll
      for (int mt = 0; mt < MT; ++mt)
        c[mt].v = wmma32(lds_load_a(At + mt * 16 * 128, 64, kc, lane), b, c[mt].v);
    }
    const int nc = nt * 16 + nl;
    const float bias = bn1[nc];
#pragma unroll
    for (int mt = 0; mt < MT; ++mt)
#pragma unroll
      for (int v = 0; v < 8; ++v) {
        float x = c[mt].f[v] + bias;
        Ht[(mt * 16 + v + half * 8) * 128 + nc] = (_Float16)fmaxf(x, 0.0f);
      }
  }
  __syncthreads();

  for (int nt = 0; nt < 4; ++nt) {
    CF c[MT];
#pragma unroll
    for (int mt = 0; mt < MT; ++mt)
#pragma unroll
      for (int v = 0; v < 8; ++v) c[mt].f[v] = 0.0f;
#pragma unroll
    for (int kc = 0; kc < 4; ++kc) {
      v16h b = glob_load_b(Wn2p, 64, kc, nt, lane);
#pragma unroll
      for (int mt = 0; mt < MT; ++mt)
        c[mt].v = wmma32(lds_load_a(Ht + mt * 16 * 128, 64, kc, lane), b, c[mt].v);
    }
    const int nc = nt * 16 + nl;
    const float bias = bn2[nc];
#pragma unroll
    for (int mt = 0; mt < MT; ++mt)
#pragma unroll
      for (int v = 0; v < 8; ++v) {
        int node = n0 + mt * 16 + v + half * 8;
        if (node < N) out[(size_t)node * 64 + nc] = c[mt].f[v] + bias;
      }
  }
}

// ---------------------------------------------------------------------------
extern "C" void kernel_launch(void* const* d_in, const int* in_sizes, int n_in,
                              void* d_out, int out_size, void* d_ws, size_t ws_size,
                              hipStream_t stream) {
  (void)n_in; (void)out_size; (void)ws_size;
  const float* z_h  = (const float*)d_in[0];
  const float* posw = (const float*)d_in[1];
  const int*   ei   = (const int*)d_in[2];
  const float* We1  = (const float*)d_in[3];
  const float* be1  = (const float*)d_in[4];
  const float* We2  = (const float*)d_in[5];
  const float* be2  = (const float*)d_in[6];
  const float* Wg1  = (const float*)d_in[7];
  const float* bg1  = (const float*)d_in[8];
  const float* Wg2  = (const float*)d_in[9];
  const float* bg2  = (const float*)d_in[10];
  const float* Wn1  = (const float*)d_in[11];
  const float* bn1  = (const float*)d_in[12];
  const float* Wn2  = (const float*)d_in[13];
  const float* bn2  = (const float*)d_in[14];
  const float* Ww1  = (const float*)d_in[15];
  const float* bw1  = (const float*)d_in[16];
  const float* Ww2  = (const float*)d_in[17];
  const float* bw2  = (const float*)d_in[18];
  float* out = (float*)d_out;

  const int F = 64;
  const int N = in_sizes[0] / F;   // 50000
  const int E = in_sizes[2] / 2;   // 800000

  char* ws = (char*)d_ws;
  size_t off = 0;
  auto take = [&](size_t bytes) -> char* {
    char* p = ws + off;
    off += (bytes + 255) & ~(size_t)255;
    return p;
  };
  _Float16* We1p  = (_Float16*)take(160 * 128 * sizeof(_Float16));
  _Float16* Wg1p  = (_Float16*)take(160 * 128 * sizeof(_Float16));
  _Float16* We2p  = (_Float16*)take(128 * 64 * sizeof(_Float16));
  _Float16* Wg2p  = (_Float16*)take(128 * 16 * sizeof(_Float16));
  _Float16* Wn1p  = (_Float16*)take(128 * 128 * sizeof(_Float16));
  _Float16* Wn2p  = (_Float16*)take(128 * 64 * sizeof(_Float16));
  _Float16* Ww1p  = (_Float16*)take(96 * 128 * sizeof(_Float16));
  _Float16* Ww2p  = (_Float16*)take(128 * 64 * sizeof(_Float16));
  float* m_agg = (float*)take((size_t)N * 64 * sizeof(float));
  float* wsum  = (float*)take(64 * sizeof(float));

  auto g = [](int n) { return (n + 255) / 256; };
  gnn_pack_w<<<g(160 * 128), 256, 0, stream>>>(We1, We1p, 136, 160, 128);
  gnn_pack_w<<<g(160 * 128), 256, 0, stream>>>(Wg1, Wg1p, 136, 160, 128);
  gnn_pack_w<<<g(128 * 64),  256, 0, stream>>>(We2, We2p, 128, 128, 64);
  gnn_pack_wvec<<<g(128 * 16), 256, 0, stream>>>(Wg2, Wg2p, 128);
  gnn_pack_w<<<g(128 * 128), 256, 0, stream>>>(Wn1, Wn1p, 128, 128, 128);
  gnn_pack_w<<<g(128 * 64),  256, 0, stream>>>(Wn2, Wn2p, 128, 128, 64);
  gnn_pack_w<<<g(96 * 128),  256, 0, stream>>>(Ww1, Ww1p, 67, 96, 128);
  gnn_pack_w<<<g(128 * 64),  256, 0, stream>>>(Ww2, Ww2p, 128, 128, 64);

  gnn_zero_f32<<<512, 256, 0, stream>>>(m_agg, N * 64);
  gnn_zero_f32<<<1, 64, 0, stream>>>(wsum, 64);

  gnn_edge_kernel<<<(E + 63) / 64, 32, 0, stream>>>(
      z_h, ei, We1p, be1, We2p, be2, Wg1p, bg1, Wg2p, bg2, m_agg, E);
  gnn_world_kernel<<<(N + 63) / 64, 32, 0, stream>>>(
      z_h, posw, Ww1p, bw1, Ww2p, bw2, wsum, N);
  gnn_node_kernel<<<(N + 63) / 64, 32, 0, stream>>>(
      z_h, m_agg, wsum, Wn1p, bn1, Wn2p, bn2, out, N);
}
